// UpBlock_5549097746512
// MI455X (gfx1250) — compile-verified, matrix operand order checked by reading, not run on hardware
//
#include <hip/hip_runtime.h>

typedef __attribute__((ext_vector_type(16))) _Float16 v16h;
typedef __attribute__((ext_vector_type(8)))  _Float16 v8h;
typedef __attribute__((ext_vector_type(8)))  float    v8f;

#define SLOPE 0.01f
#define EPSV  1e-5f

// ---------------------------------------------------------------- CDNA5 async global->LDS (one b128 per lane)
__device__ __forceinline__ void async_load_b128(unsigned lds_off, const void* gaddr) {
    asm volatile("global_load_async_to_lds_b128 %0, %1, off"
                 :: "v"(lds_off), "v"((unsigned long long)(size_t)gaddr)
                 : "memory");
}
__device__ __forceinline__ void wait_async0() {
    asm volatile("s_wait_asynccnt 0x0" ::: "memory");
}

typedef __attribute__((address_space(3))) unsigned int lds_uint_t;
__device__ __forceinline__ unsigned lds_offset_of(unsigned int* p) {
    return (unsigned)(size_t)(lds_uint_t*)p;   // addrspacecast generic -> LDS gives byte offset
}

// ---------------------------------------------------------------- prep: f32 -> f16
__global__ void k_f32_to_f16(const float* __restrict__ in, _Float16* __restrict__ out, int n) {
    int i = blockIdx.x * blockDim.x + threadIdx.x;
    int stride = gridDim.x * blockDim.x;
    for (; i < n; i += stride) out[i] = (_Float16)in[i];
}

// ---------------------------------------------------------------- prep: pack weights into WMMA B-fragment layout
// u32 index: (((tap*nkc + kc)*4 + nt)*256) + c*128 + lane*4 + j
//   v = c*4 + j, half = lane>>4, n = nt*16 + (lane&15)
//   k0 = kc*32 + ((v&4)<<2) + half*8 + (v&3)*2 ; u32 packs {W[k0,n], W[k0+1,n]}
__global__ void k_pack_w(const float* __restrict__ W, unsigned int* __restrict__ frag,
                         int K, int Cin) {
    int nkc = Cin >> 5;
    int total = K * nkc * 1024;
    int stride = gridDim.x * blockDim.x;
    for (int tid = blockIdx.x * blockDim.x + threadIdx.x; tid < total; tid += stride) {
        int j    = tid & 3;
        int lane = (tid >> 2) & 31;
        int c    = (tid >> 7) & 1;
        int nt   = (tid >> 8) & 3;
        int rest = tid >> 10;
        int kc   = rest % nkc;
        int tap  = rest / nkc;
        int v    = c * 4 + j;
        int half = lane >> 4;
        int n    = nt * 16 + (lane & 15);
        int k0   = kc * 32 + ((v & 4) << 2) + half * 8 + (v & 3) * 2;
        union { _Float16 h[2]; unsigned int u; } p;
        p.h[0] = (_Float16)W[(size_t)(tap * Cin + k0) * 64 + n];
        p.h[1] = (_Float16)W[(size_t)(tap * Cin + k0 + 1) * 64 + n];
        frag[tid] = p.u;
    }
}

// ---------------------------------------------------------------- gather-GEMM conv
// One wave = MT x (16-row x 64-col) tiles. Per-tap B fragments staged block-wide
// into LDS via async global->LDS, consumed via ds_load_b128.
// EPI==0: leaky_relu -> out32 ; EPI==1: + skip -> out16 (f16)
template<int NKC, int EPI>
__global__ void __launch_bounds__(256)
k_conv_wmma(const _Float16* __restrict__ feats, const int* __restrict__ nbr,
            const unsigned int* __restrict__ wfrag, const float* __restrict__ skip,
            float* __restrict__ out32, _Float16* __restrict__ out16,
            int Nout, int K) {
    constexpr int CIN = NKC * 32;
    constexpr int MT  = 2;                        // M-tiles per wave (32 rows)
    __shared__ unsigned int smemW[NKC * 1024];    // one tap's B fragments (NKC*4KB)

    const int tid   = threadIdx.x;
    const int lane  = tid & 31;
    const int wib   = tid >> 5;                   // wave in block: 0..7
    const int ntile = Nout >> 4;                  // Nout is a multiple of 16 here
    const int mrow  = lane & 15;
    const int half  = lane >> 4;

    const int tbase = blockIdx.x * 8 * MT + wib * MT;
    int  tile[MT]; bool valid[MT]; int rowb[MT];
#pragma unroll
    for (int mt = 0; mt < MT; ++mt) {
        int t = tbase + mt;
        valid[mt] = (t < ntile);
        tile[mt]  = valid[mt] ? t : (ntile - 1);  // clamp: compute dummy, skip store
        rowb[mt]  = tile[mt] * 16 + mrow;
    }

    v8f acc[MT][4] = {};
    const unsigned lds_base = lds_offset_of(&smemW[0]);

    for (int tap = 0; tap < K; ++tap) {
        // ---- stage this tap's B fragments: 256 threads x NKC async b128 ops
        const unsigned int* __restrict__ wtap = wfrag + (size_t)tap * (NKC * 1024);
#pragma unroll
        for (int i = 0; i < NKC; ++i) {
            const int e = tid + i * 256;          // uint4 element id
            async_load_b128(lds_base + (unsigned)e * 16u,
                            (const char*)wtap + (size_t)e * 16);
        }
        wait_async0();
        __syncthreads();

        // ---- gathered row indices for this tap
        int idx[MT];
#pragma unroll
        for (int mt = 0; mt < MT; ++mt) idx[mt] = nbr[rowb[mt] * K + tap];

#pragma unroll
        for (int kc = 0; kc < NKC; ++kc) {
            // A fragments: lanes 0-15 hold K={0..7,16..23}, lanes 16-31 K={8..15,24..31}
            v16h a[MT];
#pragma unroll
            for (int mt = 0; mt < MT; ++mt) {
                const _Float16* arow = feats + (size_t)idx[mt] * CIN + kc * 32 + half * 8;
                v8h alo = *reinterpret_cast<const v8h*>(arow);
                v8h ahi = *reinterpret_cast<const v8h*>(arow + 16);
#pragma unroll
                for (int i = 0; i < 8; ++i) { a[mt][i] = alo[i]; a[mt][i + 8] = ahi[i]; }
            }
            const uint4* qs = reinterpret_cast<const uint4*>(smemW + kc * 1024);
#pragma unroll
            for (int nt = 0; nt < 4; ++nt) {
                union { uint4 u; v8h h; } c0, c1;
                c0.u = qs[nt * 64 + lane];        // ds_load_b128
                c1.u = qs[nt * 64 + 32 + lane];
                v16h b;
#pragma unroll
                for (int i = 0; i < 8; ++i) { b[i] = c0.h[i]; b[i + 8] = c1.h[i]; }
#pragma unroll
                for (int mt = 0; mt < MT; ++mt)
                    acc[mt][nt] = __builtin_amdgcn_wmma_f32_16x16x32_f16(
                        false, a[mt], false, b, (short)0, acc[mt][nt], false, false);
            }
        }
        __syncthreads();                          // protect LDS before next tap's staging
    }

    // ---- epilogue: C/D layout -> col N = (lane&15)+16*nt, row M = r + 8*half
#pragma unroll
    for (int mt = 0; mt < MT; ++mt) {
        if (!valid[mt]) continue;
#pragma unroll
        for (int nt = 0; nt < 4; ++nt) {
            const int col = nt * 16 + mrow;
#pragma unroll
            for (int r = 0; r < 8; ++r) {
                const int row = tile[mt] * 16 + half * 8 + r;
                float v = acc[mt][nt][r];
                const size_t o = (size_t)row * 64 + col;
                if (EPI == 1) { v += skip[o]; out16[o] = (_Float16)v; }
                else          { v = v > 0.0f ? v : SLOPE * v; out32[o] = v; }
            }
        }
    }
}

// ---------------------------------------------------------------- BN stats: per-channel sum / sumsq
__global__ void k_zero(float* __restrict__ p, int n) {
    int i = blockIdx.x * blockDim.x + threadIdx.x;
    if (i < n) p[i] = 0.0f;
}

__global__ void k_reduce_stats(const float* __restrict__ y, float* __restrict__ stats, int N) {
    const int c   = threadIdx.x & 63;
    const int rg  = threadIdx.x >> 6;
    const int rpb = blockDim.x >> 6;
    float s = 0.0f, s2 = 0.0f;
    for (int r = blockIdx.x * rpb + rg; r < N; r += gridDim.x * rpb) {
        float v = y[(size_t)r * 64 + c];
        s += v; s2 += v * v;
    }
    atomicAdd(&stats[c], s);
    atomicAdd(&stats[64 + c], s2);
}

// ---------------------------------------------------------------- BN apply: (y-mean)*rsqrt(var+eps)*g+b
__global__ void k_bn_apply(const float* __restrict__ y, const float* __restrict__ stats,
                           const float* __restrict__ g, const float* __restrict__ b,
                           _Float16* __restrict__ out16, float* __restrict__ out32, int N) {
    const int total = N * 64;
    const float invN = 1.0f / (float)N;
    int i = blockIdx.x * blockDim.x + threadIdx.x;
    const int stride = gridDim.x * blockDim.x;
    for (; i < total; i += stride) {
        const int c = i & 63;
        const float mean = stats[c] * invN;
        const float var  = stats[64 + c] * invN - mean * mean;
        const float v = (y[i] - mean) * rsqrtf(var + EPSV) * g[c] + b[c];
        if (out16) out16[i] = (_Float16)v;
        if (out32) out32[i] = v;
    }
}

// ================================================================ driver
extern "C" void kernel_launch(void* const* d_in, const int* in_sizes, int n_in,
                              void* d_out, int out_size, void* d_ws, size_t ws_size,
                              hipStream_t stream) {
    const float* x_feats = (const float*)d_in[0];
    const float* skip    = (const float*)d_in[1];
    const int* nbr1   = (const int*)d_in[2];
    const int* nbr_up = (const int*)d_in[3];
    const int* nbr2   = (const int*)d_in[4];
    const int* nbr3   = (const int*)d_in[5];
    const int* nbr4   = (const int*)d_in[6];
    const float* W1  = (const float*)d_in[7];
    const float* Wup = (const float*)d_in[8];
    const float* W2  = (const float*)d_in[9];
    const float* W3  = (const float*)d_in[10];
    const float* W4  = (const float*)d_in[11];
    const float* g1 = (const float*)d_in[12]; const float* b1 = (const float*)d_in[13];
    const float* g2 = (const float*)d_in[14]; const float* b2 = (const float*)d_in[15];
    const float* g3 = (const float*)d_in[16]; const float* b3 = (const float*)d_in[17];
    const float* g4 = (const float*)d_in[18]; const float* b4 = (const float*)d_in[19];

    const int Ci = 128, Co = 64;
    const int N_low = in_sizes[0] / Ci;
    const int N_up  = in_sizes[1] / Co;

    // carve workspace
    char* p = (char*)d_ws;
    auto carve = [&](size_t bytes) -> void* {
        void* r = (void*)p;
        p += (bytes + 255) & ~(size_t)255;
        return r;
    };
    _Float16* F16X = (_Float16*)carve((size_t)N_low * Ci * 2);   // x_feats in f16
    _Float16* HA   = (_Float16*)carve((size_t)N_up * Co * 2);    // f16 activation ping
    _Float16* HB   = (_Float16*)carve((size_t)N_up * Co * 2);    // f16 activation pong
    float*    Y    = (float*)carve((size_t)N_up * Co * 4);       // raw f32 conv out (reused)
    unsigned* F1 = (unsigned*)carve((size_t)27 * 4 * 1024 * 4);  // W1 frags (Ci=128 -> nkc=4)
    unsigned* FU = (unsigned*)carve((size_t)27 * 2 * 1024 * 4);
    unsigned* F2 = (unsigned*)carve((size_t)9  * 2 * 1024 * 4);
    unsigned* F3 = (unsigned*)carve((size_t)9  * 2 * 1024 * 4);
    unsigned* F4 = (unsigned*)carve((size_t)27 * 2 * 1024 * 4);
    float* stats = (float*)carve(128 * 4);

    // --- prep
    k_f32_to_f16<<<4096, 256, 0, stream>>>(x_feats, F16X, N_low * Ci);
    k_pack_w<<<(27 * 4 * 1024 + 255) / 256, 256, 0, stream>>>(W1,  F1, 27, 128);
    k_pack_w<<<(27 * 2 * 1024 + 255) / 256, 256, 0, stream>>>(Wup, FU, 27, 64);
    k_pack_w<<<(9  * 2 * 1024 + 255) / 256, 256, 0, stream>>>(W2,  F2, 9,  64);
    k_pack_w<<<(9  * 2 * 1024 + 255) / 256, 256, 0, stream>>>(W3,  F3, 9,  64);
    k_pack_w<<<(27 * 2 * 1024 + 255) / 256, 256, 0, stream>>>(W4,  F4, 27, 64);

    const int tl = N_low >> 4, tu = N_up >> 4;
    const int gl = (tl + 15) / 16;     // 8 waves x 2 M-tiles per block
    const int gu = (tu + 15) / 16;

    // --- stage 1: conv1 (Ci=128,K=27) + leaky -> Y ; BN -> HA(f16)
    k_conv_wmma<4, 0><<<gl, 256, 0, stream>>>(F16X, nbr1, F1, nullptr, Y, nullptr, N_low, 27);
    k_zero<<<1, 128, 0, stream>>>(stats, 128);
    k_reduce_stats<<<256, 256, 0, stream>>>(Y, stats, N_low);
    k_bn_apply<<<4096, 256, 0, stream>>>(Y, stats, g1, b1, HA, nullptr, N_low);

    // --- stage 2: inverse conv (K=27) + skip -> HB(f16) (no BN/act)
    k_conv_wmma<2, 1><<<gu, 256, 0, stream>>>(HA, nbr_up, FU, skip, nullptr, HB, N_up, 27);

    // --- stage 3: conv2 (K=9) + leaky -> Y ; BN -> HA
    k_conv_wmma<2, 0><<<gu, 256, 0, stream>>>(HB, nbr2, F2, nullptr, Y, nullptr, N_up, 9);
    k_zero<<<1, 128, 0, stream>>>(stats, 128);
    k_reduce_stats<<<256, 256, 0, stream>>>(Y, stats, N_up);
    k_bn_apply<<<4096, 256, 0, stream>>>(Y, stats, g2, b2, HA, nullptr, N_up);

    // --- stage 4: conv3 (K=9) + leaky -> Y ; BN -> HB
    k_conv_wmma<2, 0><<<gu, 256, 0, stream>>>(HA, nbr3, F3, nullptr, Y, nullptr, N_up, 9);
    k_zero<<<1, 128, 0, stream>>>(stats, 128);
    k_reduce_stats<<<256, 256, 0, stream>>>(Y, stats, N_up);
    k_bn_apply<<<4096, 256, 0, stream>>>(Y, stats, g3, b3, HB, nullptr, N_up);

    // --- stage 5: conv4 (K=27) + leaky -> Y ; BN -> d_out (f32)
    k_conv_wmma<2, 0><<<gu, 256, 0, stream>>>(HB, nbr4, F4, nullptr, Y, nullptr, N_up, 27);
    k_zero<<<1, 128, 0, stream>>>(stats, 128);
    k_reduce_stats<<<256, 256, 0, stream>>>(Y, stats, N_up);
    k_bn_apply<<<4096, 256, 0, stream>>>(Y, stats, g4, b4, nullptr, (float*)d_out, N_up);
}
